// RGCN_37958920962653
// MI455X (gfx1250) — matrix operand order repaired and checked by reading, not compiled, not run on hardware
//
#include <hip/hip_runtime.h>

// ---------------- problem constants (match reference) ----------------
static constexpr int N_NODES = 50000;
static constexpr int N_EDGES = 600000;
static constexpr int IN_CH   = 128;
static constexpr int HID     = 256;
static constexpr int OUT_CH  = 64;
static constexpr int N_REL   = 8;

// ---------------- CDNA5 WMMA types ----------------
typedef __attribute__((ext_vector_type(16))) __bf16 v16bf;
typedef __attribute__((ext_vector_type(8)))  float  v8f;

__device__ __forceinline__ unsigned short f2bf(float f) {
    // round-to-nearest-even f32 -> bf16
    unsigned u = __float_as_uint(f);
    unsigned r = u + 0x7FFFu + ((u >> 16) & 1u);
    return (unsigned short)(r >> 16);
}

// ---------------- tiny utility kernels ----------------
__global__ void rgcn_fill_f32(float* __restrict__ p, float v, int n) {
    int i = blockIdx.x * blockDim.x + threadIdx.x;
    if (i < n) p[i] = v;
}
__global__ void rgcn_fill_i32(int* __restrict__ p, int v, int n) {
    int i = blockIdx.x * blockDim.x + threadIdx.x;
    if (i < n) p[i] = v;
}
__global__ void rgcn_bias_init(float* __restrict__ out, const float* __restrict__ b,
                               int H, int total) {
    int i = blockIdx.x * blockDim.x + threadIdx.x;
    if (i < total) out[i] = b[i & (H - 1)];   // H is 256 or 64 (pow2)
}
__global__ void rgcn_relu(float* __restrict__ p, int n) {
    int i = blockIdx.x * blockDim.x + threadIdx.x;
    if (i < n) p[i] = fmaxf(p[i], 0.0f);
}
__global__ void rgcn_cvt_bf16(const float* __restrict__ in,
                              unsigned short* __restrict__ out, int n) {
    int i = blockIdx.x * blockDim.x + threadIdx.x;
    if (i < n) out[i] = f2bf(in[i]);
}
// agg f32 -> bf16 with fused per-row mean scale; grid (N, K/128), block 128
__global__ void rgcn_cvt_mean(const float* __restrict__ agg, const int* __restrict__ cnt,
                              unsigned short* __restrict__ out, int K) {
    int row = blockIdx.x;
    int c   = blockIdx.y * 128 + threadIdx.x;
    float s = 1.0f / (float)max(cnt[row], 1);
    out[(size_t)row * K + c] = f2bf(agg[(size_t)row * K + c] * s);
}

// ---------------- degree count per (relation, dst) ----------------
__global__ void rgcn_count(const int* __restrict__ dst, const int* __restrict__ et,
                           int* __restrict__ cnt) {
    int e = blockIdx.x * blockDim.x + threadIdx.x;
    if (e < N_EDGES) atomicAdd(&cnt[et[e] * N_NODES + dst[e]], 1);
}

// ---------------- masked scatter-add: agg[dst] += x[src] for edges of `rel` ----
__global__ void rgcn_scatter(const float* __restrict__ xin, const int* __restrict__ src,
                             const int* __restrict__ dst, const int* __restrict__ et,
                             float* __restrict__ agg, int K, int rel) {
    int e = blockIdx.x;
    if (et[e] != rel) return;                // whole block exits together
    int s = src[e], d = dst[e];
    for (int c = threadIdx.x; c < K; c += blockDim.x)
        unsafeAtomicAdd(&agg[(size_t)d * K + c], xin[(size_t)s * K + c]);
}

// ---- pack f32 W[K x H] into per-lane WMMA B fragments (bf16 pairs in u32) ----
// one wave per (n_tile, k_chunk); fragment = 8 consecutive u32 per lane:
// u[v] = { W[k0+16*hi+2v][n], W[k0+16*hi+2v+1][n] }  with n = nt*16 + (lane&15)
__global__ void rgcn_pack_w(const float* __restrict__ W, unsigned* __restrict__ Bpk,
                            int K, int H) {
    int tile = blockIdx.x * 4 + (threadIdx.x >> 5);
    int ntiles = (H / 16) * (K / 32);
    if (tile >= ntiles) return;
    int kchunks = K / 32;
    int nt = tile / kchunks, kc = tile % kchunks;
    int lane = threadIdx.x & 31;
    int ncol = lane & 15, hi = lane >> 4;
    int n = nt * 16 + ncol, k0 = kc * 32;
    unsigned* outp = Bpk + ((size_t)tile * 32 + lane) * 8;
#pragma unroll
    for (int v = 0; v < 8; ++v) {
        int kb = k0 + 16 * hi + 2 * v;
        unsigned lo = f2bf(W[(size_t)kb * H + n]);
        unsigned up = f2bf(W[(size_t)(kb + 1) * H + n]);
        outp[v] = lo | (up << 16);
    }
}

// ---------------- bf16 WMMA GEMM: C[N x H] += A[N x K] @ W[K x H] -------------
// A: bf16 row-major. Bpk: fragment-packed bf16 weights. C: f32, accum in place.
// block = 128 threads = 4 waves; block owns a 16(M) x 64(N) C tile, wave w owns
// the 16x16 sub-tile at n-tile (blockIdx.y*4 + w). K fully unrolled -> K/32 WMMAs.
template <int K, int H>
__global__ void __launch_bounds__(128)
rgcn_gemm_wmma(const unsigned short* __restrict__ Abf,
               const unsigned* __restrict__ Bpk,
               float* __restrict__ C) {
    __shared__ unsigned short As[16 * K];    // 16-row A strip, row-major
    const int tid  = threadIdx.x;
    const int lane = tid & 31;
    const int wave = tid >> 5;
    const int m0   = blockIdx.x * 16;
    const int n0t  = blockIdx.y * 4 + wave;  // global 16-wide n-tile index
    const int ncol = lane & 15, hi = lane >> 4;

    // cooperative A-strip copy: strip is contiguous in global -> flat b128 copy
    {
        const uint4* srcp = (const uint4*)(Abf + (size_t)m0 * K);
        uint4* dstp = (uint4*)As;
        constexpr int NQ = 16 * K / 8;       // uint4 count (256 or 512)
#pragma unroll
        for (int e = tid; e < NQ; e += 128) dstp[e] = srcp[e];
    }

    // load current C tile (accumulate-in-place across the 9 GEMMs per layer)
    v8f acc;
    float* Cp = C + (size_t)m0 * H + n0t * 16 + ncol;
#pragma unroll
    for (int i = 0; i < 8; ++i) acc[i] = Cp[(size_t)(i + 8 * hi) * H];

    __syncthreads();

    const uint4* ap = (const uint4*)(As + ncol * K);           // lane's A row
    const uint4* bp = (const uint4*)Bpk + ((size_t)n0t * (K / 32) * 32 + lane) * 2;

#pragma unroll
    for (int kc = 0; kc < K / 32; ++kc) {
        union { v16bf v; uint4 q[2]; } af, bfr;
        af.q[0]  = ap[kc * 4 + 2 * hi];      // ka = 16*hi + {0,2,4,6} pairs
        af.q[1]  = ap[kc * 4 + 2 * hi + 1];  // ka = 16*hi + {8,10,12,14}
        bfr.q[0] = bp[kc * 64];              // kb = 16*hi + {0,2,4,6}
        bfr.q[1] = bp[kc * 64 + 1];          // kb = 16*hi + {8,10,12,14}
        acc = __builtin_amdgcn_wmma_f32_16x16x32_bf16(
                  false, af.v, false, bfr.v, (short)0, acc, false, false);
    }

#pragma unroll
    for (int i = 0; i < 8; ++i) Cp[(size_t)(i + 8 * hi) * H] = acc[i];
}

// ---------------- row-wise log_softmax over 64 logits, one wave32 per row -----
__global__ void rgcn_log_softmax(float* __restrict__ out) {
    int lane = threadIdx.x & 31;
    int wave = threadIdx.x >> 5;
    int row  = blockIdx.x * 4 + wave;
    if (row >= N_NODES) return;
    float a = out[(size_t)row * OUT_CH + lane];
    float b = out[(size_t)row * OUT_CH + lane + 32];
    float m = fmaxf(a, b);
#pragma unroll
    for (int off = 16; off > 0; off >>= 1)
        m = fmaxf(m, __shfl_xor(m, off, 32));
    float s = __expf(a - m) + __expf(b - m);
#pragma unroll
    for (int off = 16; off > 0; off >>= 1)
        s += __shfl_xor(s, off, 32);
    float l = m + __logf(s);
    out[(size_t)row * OUT_CH + lane]      = a - l;
    out[(size_t)row * OUT_CH + lane + 32] = b - l;
}

// ---------------- host-side orchestration ----------------
extern "C" void kernel_launch(void* const* d_in, const int* in_sizes, int n_in,
                              void* d_out, int out_size, void* d_ws, size_t ws_size,
                              hipStream_t stream) {
    const float* x     = (const float*)d_in[0];
    const int*   eidx  = (const int*)d_in[1];
    const int*   etype = (const int*)d_in[2];
    const float* W1    = (const float*)d_in[3];
    const float* root1 = (const float*)d_in[4];
    const float* b1    = (const float*)d_in[5];
    const float* W2    = (const float*)d_in[6];
    const float* root2 = (const float*)d_in[7];
    const float* b2    = (const float*)d_in[8];
    const float* linW  = (const float*)d_in[9];
    const float* linb  = (const float*)d_in[10];
    float* out = (float*)d_out;

    const int* src = eidx;               // edge_index[0]
    const int* dst = eidx + N_EDGES;     // edge_index[1]

    // ---- workspace carve-up (all segments 16B aligned) ----
    float* h1  = (float*)d_ws;                                        // N*HID f32
    float* h2  = h1 + (size_t)N_NODES * HID;                          // N*HID f32
    float* agg = h2 + (size_t)N_NODES * HID;                          // N*HID f32
    unsigned short* aggbf = (unsigned short*)(agg + (size_t)N_NODES * HID); // N*HID
    unsigned short* xbf   = aggbf + (size_t)N_NODES * HID;            // N*IN
    unsigned short* h1bf  = xbf + (size_t)N_NODES * IN_CH;            // N*HID
    unsigned short* h2bf  = h1bf + (size_t)N_NODES * HID;             // N*HID
    int* cnt = (int*)(h2bf + (size_t)N_NODES * HID);                  // R*N
    unsigned* root1p = (unsigned*)(cnt + N_REL * N_NODES);            // IN*HID/2 u32
    unsigned* W1p    = root1p + (size_t)IN_CH * HID / 2;              // R*IN*HID/2
    unsigned* root2p = W1p + (size_t)N_REL * IN_CH * HID / 2;         // HID*HID/2
    unsigned* W2p    = root2p + (size_t)HID * HID / 2;                // R*HID*HID/2
    unsigned* linWp  = W2p + (size_t)N_REL * HID * HID / 2;           // HID*OUT/2

    const int NH = N_NODES * HID;                     // 12.8M
    const int NI = N_NODES * IN_CH;                   // 6.4M
    dim3 grid_hid(N_NODES / 16, HID / 64);
    dim3 grid_out(N_NODES / 16, OUT_CH / 64);

    // ---- pack all weights into WMMA B-fragment order (bf16), once per call ----
    {
        int t1 = (HID / 16) * (IN_CH / 32);           // 64 tiles (K=128,H=256)
        int t2 = (HID / 16) * (HID / 32);             // 128 tiles (K=256,H=256)
        int t3 = (OUT_CH / 16) * (HID / 32);          // 32 tiles (K=256,H=64)
        rgcn_pack_w<<<(t1 + 3) / 4, 128, 0, stream>>>(root1, root1p, IN_CH, HID);
        for (int r = 0; r < N_REL; ++r)
            rgcn_pack_w<<<(t1 + 3) / 4, 128, 0, stream>>>(
                W1 + (size_t)r * IN_CH * HID, W1p + (size_t)r * IN_CH * HID / 2, IN_CH, HID);
        rgcn_pack_w<<<(t2 + 3) / 4, 128, 0, stream>>>(root2, root2p, HID, HID);
        for (int r = 0; r < N_REL; ++r)
            rgcn_pack_w<<<(t2 + 3) / 4, 128, 0, stream>>>(
                W2 + (size_t)r * HID * HID, W2p + (size_t)r * HID * HID / 2, HID, HID);
        rgcn_pack_w<<<(t3 + 3) / 4, 128, 0, stream>>>(linW, linWp, HID, OUT_CH);
    }

    // ---- per-(relation,node) in-degree counts (shared by both layers) ----
    rgcn_fill_i32<<<(N_REL * N_NODES + 255) / 256, 256, 0, stream>>>(cnt, 0, N_REL * N_NODES);
    rgcn_count<<<(N_EDGES + 255) / 256, 256, 0, stream>>>(dst, etype, cnt);

    // ---- layer 1: h1 = relu(x@root1 + b1 + sum_r mean_r(x)@W1[r]) ----
    rgcn_cvt_bf16<<<(NI + 255) / 256, 256, 0, stream>>>(x, xbf, NI);
    rgcn_bias_init<<<(NH + 255) / 256, 256, 0, stream>>>(h1, b1, HID, NH);
    rgcn_gemm_wmma<IN_CH, HID><<<grid_hid, 128, 0, stream>>>(xbf, root1p, h1);
    for (int r = 0; r < N_REL; ++r) {
        rgcn_fill_f32<<<(NI + 255) / 256, 256, 0, stream>>>(agg, 0.0f, NI);
        rgcn_scatter<<<N_EDGES, 128, 0, stream>>>(x, src, dst, etype, agg, IN_CH, r);
        rgcn_cvt_mean<<<dim3(N_NODES, IN_CH / 128), 128, 0, stream>>>(
            agg, cnt + (size_t)r * N_NODES, aggbf, IN_CH);
        rgcn_gemm_wmma<IN_CH, HID><<<grid_hid, 128, 0, stream>>>(
            aggbf, W1p + (size_t)r * IN_CH * HID / 2, h1);
    }
    rgcn_relu<<<(NH + 255) / 256, 256, 0, stream>>>(h1, NH);
    rgcn_cvt_bf16<<<(NH + 255) / 256, 256, 0, stream>>>(h1, h1bf, NH);

    // ---- layer 2: h2 = relu(h1@root2 + b2 + sum_r mean_r(h1)@W2[r]) ----
    rgcn_bias_init<<<(NH + 255) / 256, 256, 0, stream>>>(h2, b2, HID, NH);
    rgcn_gemm_wmma<HID, HID><<<grid_hid, 128, 0, stream>>>(h1bf, root2p, h2);
    for (int r = 0; r < N_REL; ++r) {
        rgcn_fill_f32<<<(NH + 255) / 256, 256, 0, stream>>>(agg, 0.0f, NH);
        rgcn_scatter<<<N_EDGES, 128, 0, stream>>>(h1, src, dst, etype, agg, HID, r);
        rgcn_cvt_mean<<<dim3(N_NODES, HID / 128), 128, 0, stream>>>(
            agg, cnt + (size_t)r * N_NODES, aggbf, HID);
        rgcn_gemm_wmma<HID, HID><<<grid_hid, 128, 0, stream>>>(
            aggbf, W2p + (size_t)r * HID * HID / 2, h2);
    }
    rgcn_relu<<<(NH + 255) / 256, 256, 0, stream>>>(h2, NH);
    rgcn_cvt_bf16<<<(NH + 255) / 256, 256, 0, stream>>>(h2, h2bf, NH);

    // ---- classifier + log_softmax (in place in d_out) ----
    rgcn_bias_init<<<(N_NODES * OUT_CH + 255) / 256, 256, 0, stream>>>(
        out, linb, OUT_CH, N_NODES * OUT_CH);
    rgcn_gemm_wmma<HID, OUT_CH><<<grid_out, 128, 0, stream>>>(h2bf, linWp, out);
    rgcn_log_softmax<<<(N_NODES + 3) / 4, 128, 0, stream>>>(out);
}
